// GPTLanguageModel_8942121910707
// MI455X (gfx1250) — matrix operand order repaired and checked by reading, not compiled, not run on hardware
//
#include <hip/hip_runtime.h>
#include <cstdint>
#include <cstddef>

typedef unsigned short u16;
typedef __attribute__((ext_vector_type(16))) __bf16 bf16x16;
typedef __attribute__((ext_vector_type(8)))  float  f32x8;

#define NTOK   4096      // B*T
#define TSEQ   1024
#define DMODEL 768
#define NHEAD  12
#define HSZ    64
#define FFDIM  3072
#define NLAYER 12
#define VOCAB  32000

#define BM 128
#define BN 128
#define BK 32

union U16F {
  uint32_t u[8];
  uint4    q[2];
  bf16x16  v;
};

__device__ __forceinline__ u16 f2bf(float f) {
  uint32_t x = __float_as_uint(f);
  uint32_t r = x + 0x7fffu + ((x >> 16) & 1u);   // round-to-nearest-even
  return (u16)(r >> 16);
}

__device__ __forceinline__ f32x8 wmma_bf16(const U16F& a, const U16F& b, f32x8 c) {
  return __builtin_amdgcn_wmma_f32_16x16x32_bf16(false, a.v, false, b.v,
                                                 (short)0, c, false, false);
}

// async 16B global -> LDS copy (per-lane), tracked by ASYNCcnt
__device__ __forceinline__ void async_b128(const u16* g, u16* l) {
  uint32_t loff = (uint32_t)(uintptr_t)l;        // LDS aperture: low 32 bits = LDS addr
  asm volatile("global_load_async_to_lds_b128 %0, %1, off"
               :: "v"(loff), "v"((uint64_t)(uintptr_t)g)
               : "memory");
}

// ---------------------------------------------------------------- embedding
__global__ __launch_bounds__(256)
void embed_kernel(const int* __restrict__ idx, const float* __restrict__ tok,
                  const float* __restrict__ pos, float* __restrict__ x) {
  size_t i = (size_t)blockIdx.x * 256 + threadIdx.x;   // < NTOK*DMODEL
  int c = (int)(i % DMODEL);
  int m = (int)(i / DMODEL);
  int t = m & (TSEQ - 1);
  x[i] = tok[(size_t)idx[m] * DMODEL + c] + pos[(size_t)t * DMODEL + c];
}

// -------------------------------------------- weight convert+transpose f32->bf16
// dst[n*K + k] = src(k, n);  headBlocked: src is [N/64][K][64] per-head blocks
__global__ __launch_bounds__(256)
void wtcvt_kernel(const float* __restrict__ src, u16* __restrict__ dst,
                  int K, int N, int headBlocked) {
  size_t i = (size_t)blockIdx.x * 256 + threadIdx.x;
  if (i >= (size_t)K * N) return;
  int k = (int)(i % K);
  int n = (int)(i / K);
  float v = headBlocked ? src[((size_t)(n >> 6) * K + k) * 64 + (n & 63)]
                        : src[(size_t)k * N + n];
  dst[i] = f2bf(v);
}

// ---------------------------------------------------------------- layernorm -> bf16
__global__ __launch_bounds__(256)
void layernorm_bf16_kernel(const float* __restrict__ x, const float* __restrict__ g,
                           const float* __restrict__ b, u16* __restrict__ out) {
  __shared__ float red1[256], red2[256];
  int row = blockIdx.x, tid = threadIdx.x;
  const float* xr = x + (size_t)row * DMODEL;
  float s = 0.f, ss = 0.f;
  for (int c = tid; c < DMODEL; c += 256) { float v = xr[c]; s += v; ss += v * v; }
  red1[tid] = s; red2[tid] = ss; __syncthreads();
  for (int k = 128; k > 0; k >>= 1) {
    if (tid < k) { red1[tid] += red1[tid + k]; red2[tid] += red2[tid + k]; }
    __syncthreads();
  }
  float mean = red1[0] * (1.0f / DMODEL);
  float var  = red2[0] * (1.0f / DMODEL) - mean * mean;
  float rs   = rsqrtf(var + 1e-5f);
  u16* orow = out + (size_t)row * DMODEL;
  for (int c = tid; c < DMODEL; c += 256)
    orow[c] = f2bf((xr[c] - mean) * rs * g[c] + b[c]);
}

// ---------------------------------------------------------------- WMMA GEMM
// C[M,N] = A[M,K](bf16,row-major) * Bt[N,K](bf16, pre-transposed weights).
// 256 threads = 8 waves; 128x128 block tile; 64x32 per wave; async double-buffered
// LDS staging. Epilogue storeMode:
//  0: f32 row-major            1: f32 row-major + resid add
//  2: bf16 row-major           3: bf16 scattered [b,h,t,hs] (q/k)
//  4: bf16 scattered [b,h,hs,t] (v transposed)
__global__ __launch_bounds__(256)
void gemm_bf16_kernel(const u16* __restrict__ A, const u16* __restrict__ Bt,
                      const float* __restrict__ bias, const float* __restrict__ resid,
                      void* __restrict__ out,
                      int M, int N, int K, int storeMode, int doRelu) {
  __shared__ u16 Asm[2][BM * BK];
  __shared__ u16 Bsm[2][BN * BK];
  const int tid  = threadIdx.x;
  const int lane = tid & 31, wv = tid >> 5;
  const int hi   = lane >> 4, ln = lane & 15;
  const int mblk = blockIdx.y, nblk = blockIdx.x;
  const int wr   = (wv >> 2) * 64;     // 0 / 64
  const int wc   = (wv & 3) * 32;      // 0 / 32 / 64 / 96

  const int kTiles = K >> 5;

  // issue async copies for one K-tile (4 x b128 per thread = full 8KB+8KB tile)
  auto issue = [&](int kt, int buf) {
#pragma unroll
    for (int v = 0; v < 2; ++v) {
      int c   = tid * 2 + v;           // 0..511
      int row = c >> 2, seg = c & 3;   // 4 x 16B per 32-elem row
      async_b128(A  + (size_t)(mblk * BM + row) * K + kt * BK + seg * 8,
                 &Asm[buf][row * BK + seg * 8]);
      async_b128(Bt + (size_t)(nblk * BN + row) * K + kt * BK + seg * 8,
                 &Bsm[buf][row * BK + seg * 8]);
    }
  };

  f32x8 acc[4][2] = {};
  issue(0, 0);

  for (int kt = 0; kt < kTiles; ++kt) {
    if (kt + 1 < kTiles) {
      issue(kt + 1, (kt + 1) & 1);
      asm volatile("s_wait_asynccnt 0x4" ::: "memory");  // tile kt landed
    } else {
      asm volatile("s_wait_asynccnt 0x0" ::: "memory");
    }
    __syncthreads();

    const int buf = kt & 1;
    const uint32_t* aw = (const uint32_t*)Asm[buf];
    const uint32_t* bw = (const uint32_t*)Bsm[buf];
    U16F af[4], bfr[2];
#pragma unroll
    for (int i = 0; i < 4; ++i) {        // A frag: 16 rows x 32 K
      int base = (wr + i * 16 + ln) * 16;
#pragma unroll
      for (int e2 = 0; e2 < 4; ++e2) {
        af[i].u[e2]     = aw[base + 8 * hi + e2];       // K 0-7 / 16-23
        af[i].u[4 + e2] = aw[base + 8 * hi + 4 + e2];   // K 8-15 / 24-31
      }
    }
#pragma unroll
    for (int j = 0; j < 2; ++j) {        // B frag: 32 K x 16 cols
      int base = (wc + j * 16 + ln) * 16 + 8 * hi;
#pragma unroll
      for (int e2 = 0; e2 < 8; ++e2) bfr[j].u[e2] = bw[base + e2];
    }
#pragma unroll
    for (int i = 0; i < 4; ++i)
#pragma unroll
      for (int j = 0; j < 2; ++j)
        acc[i][j] = wmma_bf16(af[i], bfr[j], acc[i][j]);
    __syncthreads();                     // tile fully consumed before reuse
  }

#pragma unroll
  for (int i = 0; i < 4; ++i) {
#pragma unroll
    for (int j = 0; j < 2; ++j) {
#pragma unroll
      for (int r = 0; r < 8; ++r) {
        int row = mblk * BM + wr + i * 16 + r + 8 * hi;
        int col = nblk * BN + wc + j * 16 + ln;
        float val = acc[i][j][r];
        if (bias)   val += bias[col];
        if (doRelu) val = fmaxf(val, 0.0f);
        if (storeMode == 0) {
          ((float*)out)[(size_t)row * N + col] = val;
        } else if (storeMode == 1) {
          ((float*)out)[(size_t)row * N + col] = val + resid[(size_t)row * N + col];
        } else if (storeMode == 2) {
          ((u16*)out)[(size_t)row * N + col] = f2bf(val);
        } else if (storeMode == 3) {
          int b = row >> 10, t = row & 1023, h = col >> 6, hs = col & 63;
          ((u16*)out)[(((size_t)b * NHEAD + h) * TSEQ + t) * HSZ + hs] = f2bf(val);
        } else {
          int b = row >> 10, t = row & 1023, h = col >> 6, hs = col & 63;
          ((u16*)out)[(((size_t)b * NHEAD + h) * HSZ + hs) * TSEQ + t] = f2bf(val);
        }
      }
    }
  }
}

// ---------------------------------------------------------------- flash attention
// q,k: [B,H,T,64] bf16; vT: [B,H,64,T] bf16; out: [B,T,H*64] bf16.
// One wave owns 16 query rows, streams 32-key blocks with online softmax.
__global__ __launch_bounds__(128)
void attn_kernel(const u16* __restrict__ q, const u16* __restrict__ k,
                 const u16* __restrict__ vT, u16* __restrict__ attout) {
  const int lane = threadIdx.x & 31;
  const int wv   = threadIdx.x >> 5;
  const int hi   = lane >> 4, ln = lane & 15;
  const int bh   = blockIdx.y;                 // 0..B*H-1
  const int b    = bh / NHEAD, h = bh % NHEAD;
  const int qb0  = (blockIdx.x * 4 + wv) * 16; // query row base

  __shared__ u16 Plds[4 * 16 * 32];
  u16* myP = Plds + wv * 512;

  U16F qf[2];
  {
    const u16* qbase = q + ((size_t)bh * TSEQ + qb0 + ln) * HSZ;
#pragma unroll
    for (int c = 0; c < 2; ++c) {
      qf[c].q[0] = *(const uint4*)(qbase + c * 32 + hi * 16);
      qf[c].q[1] = *(const uint4*)(qbase + c * 32 + hi * 16 + 8);
    }
  }

  float m[8], l[8], alpha[8];
#pragma unroll
  for (int r = 0; r < 8; ++r) { m[r] = -1e30f; l[r] = 0.f; }
  f32x8 O[4] = {};

  for (int jb = 0; jb <= qb0 + 15; jb += 32) {
    f32x8 S0 = {}, S1 = {};
#pragma unroll
    for (int c = 0; c < 2; ++c) {
      U16F k0, k1;
      const u16* kb0 = k + ((size_t)bh * TSEQ + jb + ln) * HSZ + c * 32 + hi * 16;
      const u16* kb1 = k + ((size_t)bh * TSEQ + jb + 16 + ln) * HSZ + c * 32 + hi * 16;
      k0.q[0] = ((const uint4*)kb0)[0]; k0.q[1] = ((const uint4*)kb0)[1];
      k1.q[0] = ((const uint4*)kb1)[0]; k1.q[1] = ((const uint4*)kb1)[1];
      S0 = wmma_bf16(qf[c], k0, S0);
      S1 = wmma_bf16(qf[c], k1, S1);
    }
#pragma unroll
    for (int r = 0; r < 8; ++r) {
      int qrow = qb0 + r + 8 * hi;
      float s0 = S0[r] * 0.125f; if (jb + ln      > qrow) s0 = -1e30f;
      float s1 = S1[r] * 0.125f; if (jb + 16 + ln > qrow) s1 = -1e30f;
      float mx = fmaxf(s0, s1);
      for (int o = 1; o < 16; o <<= 1) mx = fmaxf(mx, __shfl_xor(mx, o, 16));
      float mn = fmaxf(m[r], mx);
      float al = expf(m[r] - mn);
      float p0 = expf(s0 - mn), p1 = expf(s1 - mn);
      float rs = p0 + p1;
      for (int o = 1; o < 16; o <<= 1) rs += __shfl_xor(rs, o, 16);
      l[r] = l[r] * al + rs;
      m[r] = mn;
      alpha[r] = al;
      myP[(r + 8 * hi) * 32 + ln]      = f2bf(p0);
      myP[(r + 8 * hi) * 32 + 16 + ln] = f2bf(p1);
    }
#pragma unroll
    for (int d = 0; d < 4; ++d)
#pragma unroll
      for (int r = 0; r < 8; ++r) O[d][r] *= alpha[r];

    asm volatile("s_wait_dscnt 0" ::: "memory");   // LDS RAW within wave

    U16F pf;
    {
      const uint32_t* pw = (const uint32_t*)myP;
      int base = ln * 16;
#pragma unroll
      for (int e2 = 0; e2 < 4; ++e2) {
        pf.u[e2]     = pw[base + 8 * hi + e2];
        pf.u[4 + e2] = pw[base + 8 * hi + 4 + e2];
      }
    }
#pragma unroll
    for (int d = 0; d < 4; ++d) {
      U16F vf;
      const u16* vb = vT + ((size_t)bh * HSZ + d * 16 + ln) * TSEQ + jb + 16 * hi;
      vf.q[0] = ((const uint4*)vb)[0];
      vf.q[1] = ((const uint4*)vb)[1];
      O[d] = wmma_bf16(pf, vf, O[d]);
    }
  }

  float inv[8];
#pragma unroll
  for (int r = 0; r < 8; ++r) inv[r] = 1.0f / l[r];
#pragma unroll
  for (int d = 0; d < 4; ++d)
#pragma unroll
    for (int r = 0; r < 8; ++r) {
      int t = qb0 + r + 8 * hi;
      attout[((size_t)b * TSEQ + t) * DMODEL + h * HSZ + d * 16 + ln] =
          f2bf(O[d][r] * inv[r]);
    }
}

// ---------------------------------------------------------------- cross entropy
__global__ __launch_bounds__(256)
void loss_row_kernel(const float* __restrict__ logits, const int* __restrict__ targets,
                     float* __restrict__ nll) {
  __shared__ float red[256];
  int row = blockIdx.x, tid = threadIdx.x;
  const float* p = logits + (size_t)row * VOCAB;
  float mx = -3.4e38f;
  for (int c = tid; c < VOCAB; c += 256) mx = fmaxf(mx, p[c]);
  red[tid] = mx; __syncthreads();
  for (int k = 128; k > 0; k >>= 1) {
    if (tid < k) red[tid] = fmaxf(red[tid], red[tid + k]);
    __syncthreads();
  }
  float M = red[0]; __syncthreads();
  float s = 0.f;
  for (int c = tid; c < VOCAB; c += 256) s += expf(p[c] - M);
  red[tid] = s; __syncthreads();
  for (int k = 128; k > 0; k >>= 1) {
    if (tid < k) red[tid] += red[tid + k];
    __syncthreads();
  }
  if (tid == 0) nll[row] = -(p[targets[row]] - M - logf(red[0]));
}

__global__ __launch_bounds__(256)
void loss_reduce_kernel(const float* __restrict__ nll, float* __restrict__ out) {
  __shared__ float red[256];
  int tid = threadIdx.x;
  float s = 0.f;
  for (int i = tid; i < NTOK; i += 256) s += nll[i];
  red[tid] = s; __syncthreads();
  for (int k = 128; k > 0; k >>= 1) {
    if (tid < k) red[tid] += red[tid + k];
    __syncthreads();
  }
  if (tid == 0) out[0] = red[0] * (1.0f / NTOK);
}

// ================================================================ host
extern "C" void kernel_launch(void* const* d_in, const int* in_sizes, int n_in,
                              void* d_out, int out_size, void* d_ws, size_t ws_size,
                              hipStream_t stream) {
  (void)in_sizes; (void)n_in; (void)out_size; (void)ws_size;
  const int*   idx     = (const int*)d_in[0];
  const int*   targets = (const int*)d_in[1];
  const float* tok_emb = (const float*)d_in[2];
  const float* pos_emb = (const float*)d_in[3];
  const float* Wq      = (const float*)d_in[4];
  const float* Wk      = (const float*)d_in[5];
  const float* Wv      = (const float*)d_in[6];
  const float* Wo      = (const float*)d_in[7];
  const float* bo      = (const float*)d_in[8];
  const float* W1      = (const float*)d_in[9];
  const float* b1      = (const float*)d_in[10];
  const float* W2      = (const float*)d_in[11];
  const float* b2      = (const float*)d_in[12];
  const float* ln1_g   = (const float*)d_in[13];
  const float* ln1_b   = (const float*)d_in[14];
  const float* ln2_g   = (const float*)d_in[15];
  const float* ln2_b   = (const float*)d_in[16];
  const float* lnf_g   = (const float*)d_in[17];
  const float* lnf_b   = (const float*)d_in[18];
  const float* Wlm     = (const float*)d_in[19];
  const float* blm     = (const float*)d_in[20];

  char* ws = (char*)d_ws;
  size_t off = 0;
  auto alloc = [&](size_t bytes) -> void* {
    off = (off + 255) & ~(size_t)255;
    void* p = ws + off; off += bytes; return p;
  };
  const size_t nDD = (size_t)DMODEL * DMODEL;   // 589824
  const size_t nDF = (size_t)DMODEL * FFDIM;    // 2359296
  const size_t nLM = (size_t)DMODEL * VOCAB;    // 24576000
  u16*   bWq  = (u16*)alloc(nDD * 2);           // all weights stored [N][K] bf16
  u16*   bWk  = (u16*)alloc(nDD * 2);
  u16*   bWv  = (u16*)alloc(nDD * 2);
  u16*   bWo  = (u16*)alloc(nDD * 2);
  u16*   bW1  = (u16*)alloc(nDF * 2);
  u16*   bW2  = (u16*)alloc(nDF * 2);
  u16*   bWlm = (u16*)alloc(nLM * 2);
  float* xbuf = (float*)alloc((size_t)NTOK * DMODEL * 4);
  u16*   hbf  = (u16*)alloc((size_t)NTOK * DMODEL * 2);
  u16*   qbf  = (u16*)alloc((size_t)NTOK * DMODEL * 2);
  u16*   kbf  = (u16*)alloc((size_t)NTOK * DMODEL * 2);
  u16*   vtbf = (u16*)alloc((size_t)NTOK * DMODEL * 2);
  u16*   atbf = (u16*)alloc((size_t)NTOK * DMODEL * 2);
  u16*   ffbf = (u16*)alloc((size_t)NTOK * FFDIM * 2);
  float* nllb = (float*)alloc((size_t)NTOK * 4);

  embed_kernel<<<(NTOK * DMODEL) / 256, 256, 0, stream>>>(idx, tok_emb, pos_emb, xbuf);
  wtcvt_kernel<<<((int)nLM + 255) / 256, 256, 0, stream>>>(Wlm, bWlm, DMODEL, VOCAB, 0);

  const dim3 gDD(DMODEL / BN, NTOK / BM);       // 6 x 32
  const dim3 gFF(FFDIM / BN, NTOK / BM);        // 24 x 32
  const dim3 gLM(VOCAB / BN, NTOK / BM);        // 250 x 32
  const dim3 gAT(TSEQ / 64, 4 * NHEAD);

  for (int l = 0; l < NLAYER; ++l) {
    const float* wq = Wq + (size_t)l * nDD;
    const float* wk = Wk + (size_t)l * nDD;
    const float* wv = Wv + (size_t)l * nDD;
    const float* wo = Wo + (size_t)l * nDD;
    const float* w1 = W1 + (size_t)l * nDF;
    const float* w2 = W2 + (size_t)l * nDF;
    wtcvt_kernel<<<((int)nDD + 255) / 256, 256, 0, stream>>>(wq, bWq, DMODEL, DMODEL, 1);
    wtcvt_kernel<<<((int)nDD + 255) / 256, 256, 0, stream>>>(wk, bWk, DMODEL, DMODEL, 1);
    wtcvt_kernel<<<((int)nDD + 255) / 256, 256, 0, stream>>>(wv, bWv, DMODEL, DMODEL, 1);
    wtcvt_kernel<<<((int)nDD + 255) / 256, 256, 0, stream>>>(wo, bWo, DMODEL, DMODEL, 0);
    wtcvt_kernel<<<((int)nDF + 255) / 256, 256, 0, stream>>>(w1, bW1, DMODEL, FFDIM, 0);
    wtcvt_kernel<<<((int)nDF + 255) / 256, 256, 0, stream>>>(w2, bW2, FFDIM, DMODEL, 0);

    layernorm_bf16_kernel<<<NTOK, 256, 0, stream>>>(xbuf, ln1_g + l * DMODEL,
                                                    ln1_b + l * DMODEL, hbf);
    gemm_bf16_kernel<<<gDD, 256, 0, stream>>>(hbf, bWq, nullptr, nullptr, qbf,
                                              NTOK, DMODEL, DMODEL, 3, 0);
    gemm_bf16_kernel<<<gDD, 256, 0, stream>>>(hbf, bWk, nullptr, nullptr, kbf,
                                              NTOK, DMODEL, DMODEL, 3, 0);
    gemm_bf16_kernel<<<gDD, 256, 0, stream>>>(hbf, bWv, nullptr, nullptr, vtbf,
                                              NTOK, DMODEL, DMODEL, 4, 0);
    attn_kernel<<<gAT, 128, 0, stream>>>(qbf, kbf, vtbf, atbf);
    gemm_bf16_kernel<<<gDD, 256, 0, stream>>>(atbf, bWo, bo + l * DMODEL, xbuf, xbuf,
                                              NTOK, DMODEL, DMODEL, 1, 0);
    layernorm_bf16_kernel<<<NTOK, 256, 0, stream>>>(xbuf, ln2_g + l * DMODEL,
                                                    ln2_b + l * DMODEL, hbf);
    gemm_bf16_kernel<<<gFF, 256, 0, stream>>>(hbf, bW1, b1 + (size_t)l * FFDIM, nullptr,
                                              ffbf, NTOK, FFDIM, DMODEL, 2, 1);
    gemm_bf16_kernel<<<gDD, 256, 0, stream>>>(ffbf, bW2, b2 + l * DMODEL, xbuf, xbuf,
                                              NTOK, DMODEL, FFDIM, 1, 0);
  }

  layernorm_bf16_kernel<<<NTOK, 256, 0, stream>>>(xbuf, lnf_g, lnf_b, hbf);
  gemm_bf16_kernel<<<gLM, 256, 0, stream>>>(hbf, bWlm, blm, nullptr, d_out,
                                            NTOK, VOCAB, DMODEL, 0, 0);

  loss_row_kernel<<<NTOK, 256, 0, stream>>>((const float*)d_out, targets, nllb);
  loss_reduce_kernel<<<1, 256, 0, stream>>>(nllb, (float*)d_out + (size_t)NTOK * VOCAB);
}
